// Neural_receiver_12086037971495
// MI455X (gfx1250) — compile-verified, matrix operand order checked by reading, not run on hardware
//
#include <hip/hip_runtime.h>
#include <math.h>

// ---------------- feature detection ----------------
#if defined(__has_builtin)
#  if __has_builtin(__builtin_amdgcn_global_load_async_to_lds_b128)
#    define HAVE_ASYNC_LDS 1
#  endif
#endif

// ---------------- types ----------------
typedef __attribute__((ext_vector_type(16))) __bf16 v16bf;
typedef __attribute__((ext_vector_type(8)))  float  v8f;
typedef __attribute__((ext_vector_type(4)))  int    v4i_t;
typedef unsigned long long ull;

// ---------------- problem constants ----------------
#define B_DIM   16
#define S_DIM   4
#define T_DIM   12
#define F_DIM   624
#define NP_DIM  7488            // T*F
#define H_DIM   256
#define TM_DIM  1024
#define CM_DIM  1024
#define L_DIM   8
#define RTOT    (B_DIM * NP_DIM)   // 119808 (= 936 * 128)
#define CHM     (RTOT / 8)         // 14976  (= 117 * 128)

// GEMM tile config: 256 threads = 8 waves; block tile 128x128; K-step 64
#define BM 128
#define BN 128
#define BK 64
#define SA 72    // LDS row stride in bf16 elems (72*2 = 144B, 16B aligned)

// epilogue modes
#define EPI_F32_BIAS    0
#define EPI_BF16_GELU   1
#define EPI_F32_ACCUM   2
#define EPI_TRANS_ACCUM 3

#define CDIV(a, b) (((a) + (b) - 1) / (b))

__device__ __forceinline__ float gelu_exact(float x) {
  return 0.5f * x * (1.0f + erff(x * 0.70710678118654752f));
}

__device__ __forceinline__ v16bf frag_ld(const __bf16* p) {
  union { uint4 u[2]; v16bf v; } t;
  t.u[0] = *reinterpret_cast<const uint4*>(p);
  t.u[1] = *reinterpret_cast<const uint4*>(p + 16);
  return t.v;
}

#ifdef HAVE_ASYNC_LDS
// per-lane 16B global -> LDS async copy (tracked by ASYNCcnt, no VGPR staging)
// builtin signature (from probe diagnostic): (v4i addrspace(1)*, v4i addrspace(3)*, Imm, Imm)
__device__ __forceinline__ void async_cp16(const __bf16* g, __bf16* l) {
  __builtin_amdgcn_global_load_async_to_lds_b128(
      (__attribute__((address_space(1))) v4i_t*)(void*)g,
      (__attribute__((address_space(3))) v4i_t*)(void*)l, 0, 0);
}
__device__ __forceinline__ void wait_async0() {
#if __has_builtin(__builtin_amdgcn_s_wait_asynccnt)
  __builtin_amdgcn_s_wait_asynccnt(0);
#else
  asm volatile("s_wait_asynccnt 0x0" ::: "memory");
#endif
}
#define GEMM_BOUNDS __launch_bounds__(256)
#else
#define GEMM_BOUNDS __launch_bounds__(256, 1)
#endif

// ---------------------------------------------------------------------------
// Tiled bf16 WMMA GEMM:  D(MxN) = A(MxK) * B(KxN), B supplied TRANSPOSED as
// Bt[N][K] row-major so both LDS fills are contiguous 16B chunks.
// M is a multiple of 128 by construction (no M guard). N guard + epilogue
// mode are template params. Double-buffered LDS; tiles staged with
// global_load_async_to_lds_b128 (s_wait_asynccnt + one barrier per K-tile).
// Batched via blockIdx.z element strides (weights: stride 0).
// ---------------------------------------------------------------------------
template <int MODE, bool NGUARD>
__global__ GEMM_BOUNDS
void gemm_bf16_wmma(const __bf16* __restrict__ A, ull strideA, int lda,
                    const __bf16* __restrict__ Bt, ull strideB, int ldb,
                    const float* __restrict__ bias,
                    void* __restrict__ Cout, ull strideC, int ldc,
                    int N, int K, int Nlog)
{
  __shared__ __align__(16) __bf16 As[2][BM * SA];
  __shared__ __align__(16) __bf16 Bs[2][BM * SA];

  const int tid = threadIdx.x;
  const int bz  = blockIdx.z;
  A  += (size_t)bz * strideA;
  Bt += (size_t)bz * strideB;

  const int n0 = blockIdx.x * BN;
  const int m0 = blockIdx.y * BM;

  // fill coords: tile = 128 rows x 64 cols bf16 = 1024 16B-chunks, 4/thread
  int frow[4], fkc[4];
  #pragma unroll
  for (int i = 0; i < 4; ++i) {
    int c = tid + (i << 8);
    frow[i] = c >> 3;
    fkc[i]  = (c & 7) << 3;
  }

  const int w    = tid >> 5;          // wave 0..7
  const int lane = tid & 31;
  const int wm   = w >> 2;            // 0..1 (64-row stripe)
  const int wn   = w & 3;             // 0..3 (32-col stripe)
  const int lrow = lane & 15;
  const int kb   = (lane >> 4) << 3;  // K-half select per ISA 16-bit layout

  v8f acc[4][2];
  #pragma unroll
  for (int i = 0; i < 4; ++i)
    #pragma unroll
    for (int j = 0; j < 2; ++j)
      acc[i][j] = (v8f){0.f, 0.f, 0.f, 0.f, 0.f, 0.f, 0.f, 0.f};

  const int nk = K / BK;

  auto compute = [&](int cur) {
    #pragma unroll
    for (int kk = 0; kk < BK; kk += 32) {
      v16bf af[4], bfv[2];
      #pragma unroll
      for (int ni = 0; ni < 2; ++ni)
        bfv[ni] = frag_ld(&Bs[cur][(wn * 32 + ni * 16 + lrow) * SA + kk + kb]);
      #pragma unroll
      for (int mi = 0; mi < 4; ++mi)
        af[mi] = frag_ld(&As[cur][(wm * 64 + mi * 16 + lrow) * SA + kk + kb]);
      #pragma unroll
      for (int mi = 0; mi < 4; ++mi)
        #pragma unroll
        for (int ni = 0; ni < 2; ++ni)
          acc[mi][ni] = __builtin_amdgcn_wmma_f32_16x16x32_bf16(
              false, af[mi], false, bfv[ni], (short)0, acc[mi][ni], false, false);
    }
  };

#ifdef HAVE_ASYNC_LDS
  // guarded tiles: OOB B rows are zeroed once in both buffers; the async
  // copies below are lane-masked so those rows are never overwritten.
  if (NGUARD) {
    #pragma unroll
    for (int i = 0; i < 4; ++i)
      if ((n0 + frow[i]) >= N) {
        uint4 z = make_uint4(0u, 0u, 0u, 0u);
        *reinterpret_cast<uint4*>(&Bs[0][frow[i] * SA + fkc[i]]) = z;
        *reinterpret_cast<uint4*>(&Bs[1][frow[i] * SA + fkc[i]]) = z;
      }
  }

  auto issue_tile = [&](int k0, int buf) {
    #pragma unroll
    for (int i = 0; i < 4; ++i) {
      async_cp16(A + (size_t)(m0 + frow[i]) * lda + k0 + fkc[i],
                 &As[buf][frow[i] * SA + fkc[i]]);
      if (!NGUARD || (n0 + frow[i]) < N)
        async_cp16(Bt + (size_t)(n0 + frow[i]) * ldb + k0 + fkc[i],
                   &Bs[buf][frow[i] * SA + fkc[i]]);
    }
  };

  issue_tile(0, 0);
  for (int kt = 0; kt < nk; ++kt) {
    const int cur = kt & 1;
    wait_async0();                      // this wave's copies into buf[cur] done
    __syncthreads();                    // everyone's copies done; prior reads retired
    if (kt + 1 < nk) issue_tile((kt + 1) * BK, cur ^ 1);
    if (kt + 2 < nk) {                  // L2 prefetch of tile after next
      const int kp = (kt + 2) * BK;
      __builtin_prefetch(A  + (size_t)(m0 + frow[0]) * lda + kp + fkc[0], 0, 1);
      __builtin_prefetch(Bt + (size_t)(n0 + frow[0]) * ldb + kp + fkc[0], 0, 1);
    }
    compute(cur);
  }
#else
  // fallback: register-staged double buffering
  uint4 pa[4], pb[4];
  #pragma unroll
  for (int i = 0; i < 4; ++i) {
    pa[i] = *reinterpret_cast<const uint4*>(A + (size_t)(m0 + frow[i]) * lda + fkc[i]);
    if (!NGUARD || (n0 + frow[i]) < N)
      pb[i] = *reinterpret_cast<const uint4*>(Bt + (size_t)(n0 + frow[i]) * ldb + fkc[i]);
    else
      pb[i] = make_uint4(0u, 0u, 0u, 0u);
  }
  #pragma unroll
  for (int i = 0; i < 4; ++i) {
    *reinterpret_cast<uint4*>(&As[0][frow[i] * SA + fkc[i]]) = pa[i];
    *reinterpret_cast<uint4*>(&Bs[0][frow[i] * SA + fkc[i]]) = pb[i];
  }
  for (int kt = 0; kt < nk; ++kt) {
    __syncthreads();
    const int  cur  = kt & 1;
    const bool more = (kt + 1) < nk;
    if (more) {
      const int k0 = (kt + 1) * BK;
      #pragma unroll
      for (int i = 0; i < 4; ++i) {
        pa[i] = *reinterpret_cast<const uint4*>(A + (size_t)(m0 + frow[i]) * lda + k0 + fkc[i]);
        if (!NGUARD || (n0 + frow[i]) < N)
          pb[i] = *reinterpret_cast<const uint4*>(Bt + (size_t)(n0 + frow[i]) * ldb + k0 + fkc[i]);
        else
          pb[i] = make_uint4(0u, 0u, 0u, 0u);
      }
    }
    compute(cur);
    if (more) {
      const int nxt = cur ^ 1;
      #pragma unroll
      for (int i = 0; i < 4; ++i) {
        *reinterpret_cast<uint4*>(&As[nxt][frow[i] * SA + fkc[i]]) = pa[i];
        *reinterpret_cast<uint4*>(&Bs[nxt][frow[i] * SA + fkc[i]]) = pb[i];
      }
    }
  }
#endif

  // epilogue: C/D layout = lane(0..15)->N, lane-half -> M 0..7 / 8..15
  const int colbase = n0 + wn * 32;
  #pragma unroll
  for (int mi = 0; mi < 4; ++mi) {
    #pragma unroll
    for (int ni = 0; ni < 2; ++ni) {
      const int col   = colbase + ni * 16 + (lane & 15);
      const int rbase = m0 + wm * 64 + mi * 16 + ((lane < 16) ? 0 : 8);
      if (NGUARD && col >= Nlog) continue;
      const float bcol = bias[col];
      #pragma unroll
      for (int r = 0; r < 8; ++r) {
        const int row = rbase + r;
        const float vv = acc[mi][ni][r] + bcol;
        if (MODE == EPI_F32_BIAS) {
          ((float*)Cout + (size_t)bz * strideC)[(size_t)row * ldc + col] = vv;
        } else if (MODE == EPI_BF16_GELU) {
          ((__bf16*)Cout + (size_t)bz * strideC)[(size_t)row * ldc + col] =
              (__bf16)gelu_exact(vv);
        } else if (MODE == EPI_F32_ACCUM) {
          ((float*)Cout + (size_t)bz * strideC)[(size_t)row * ldc + col] += vv;
        } else { // EPI_TRANS_ACCUM: x[p][h] += yv[h][p]
          ((float*)Cout + (size_t)bz * strideC)[(size_t)col * ldc + row] += vv;
        }
      }
    }
  }
}

// ---------------------------------------------------------------------------
// LayerNorm over H=256: one wave per row, wave32 shfl reduction; bf16 out,
// optionally transposed to [B, H, NP] for the token-mix GEMM.
// ---------------------------------------------------------------------------
__global__ __launch_bounds__(256)
void layernorm_bf16(const float* __restrict__ x, const float* __restrict__ g,
                    const float* __restrict__ bta, __bf16* __restrict__ out,
                    int rows, int transposed)
{
  int wv   = threadIdx.x >> 5;
  int lane = threadIdx.x & 31;
  int row  = blockIdx.x * 8 + wv;
  if (row >= rows) return;
  const float* xr = x + (size_t)row * H_DIM;
  float v[8], s = 0.f, ss = 0.f;
  #pragma unroll
  for (int i = 0; i < 8; ++i) {
    v[i] = xr[lane + i * 32];
    s += v[i]; ss += v[i] * v[i];
  }
  #pragma unroll
  for (int off = 16; off > 0; off >>= 1) {
    s  += __shfl_xor(s,  off, 32);
    ss += __shfl_xor(ss, off, 32);
  }
  float mean = s * (1.f / H_DIM);
  float var  = ss * (1.f / H_DIM) - mean * mean;
  float inv  = rsqrtf(var + 1e-5f);
  if (!transposed) {
    __bf16* o = out + (size_t)row * H_DIM;
    #pragma unroll
    for (int i = 0; i < 8; ++i) {
      int h = lane + i * 32;
      o[h] = (__bf16)((v[i] - mean) * inv * g[h] + bta[h]);
    }
  } else {
    int b = row / NP_DIM, p = row % NP_DIM;
    #pragma unroll
    for (int i = 0; i < 8; ++i) {
      int h = lane + i * 32;
      out[((size_t)b * H_DIM + h) * NP_DIM + p] =
          (__bf16)((v[i] - mean) * inv * g[h] + bta[h]);
    }
  }
}

// ---------------------------------------------------------------------------
// fp32 [K][N] -> bf16 transposed [N][K], LDS-tiled 32x32 (both sides coalesced)
// ---------------------------------------------------------------------------
__global__ __launch_bounds__(256)
void convert_transpose_bf16(const float* __restrict__ src, __bf16* __restrict__ dst,
                            int K, int N)
{
  __shared__ float tile[32][33];
  int n0 = blockIdx.x * 32;
  int k0 = blockIdx.y * 32;
  int tx = threadIdx.x & 31;
  int ty = threadIdx.x >> 5;   // 0..7
  #pragma unroll
  for (int i = 0; i < 4; ++i)
    tile[ty + i * 8][tx] = src[(size_t)(k0 + ty + i * 8) * N + n0 + tx];
  __syncthreads();
  #pragma unroll
  for (int i = 0; i < 4; ++i)
    dst[(size_t)(n0 + ty + i * 8) * K + k0 + tx] = (__bf16)tile[tx][ty + i * 8];
}

// ---------------------------------------------------------------------------
// Embed input rows: concat([y, pilot, y*scale]) as [B, NP, 24] (reference's
// C-order reshape folds s into the patch dim, 4 consecutive f per row),
// K padded 24 -> 64 with zeros, bf16.
// ---------------------------------------------------------------------------
__global__ void pack_embed_input(const float* __restrict__ y,
                                 const float* __restrict__ tp,
                                 __bf16* __restrict__ Xin, float scale)
{
  int idx = blockIdx.x * blockDim.x + threadIdx.x;
  const int total = RTOT * 64;
  if (idx >= total) return;
  int j  = idx & 63;
  int bp = idx >> 6;
  int p  = bp % NP_DIM;
  int b  = bp / NP_DIM;
  __bf16 o = (__bf16)0.f;
  if (j < 24) {
    int s   = p / 1872;
    int rem = p % 1872;
    int t   = rem / 156;
    int grp = rem % 156;
    int fl  = j / 6;
    int c   = j % 6;
    int f   = grp * 4 + fl;
    size_t base = ((((size_t)b * S_DIM + s) * T_DIM + t) * F_DIM + f) * 2;
    float val;
    if (c < 2)      val = y[base + c];
    else if (c < 4) val = tp[base + (c - 2)];
    else            val = y[base + (c - 4)] * scale;
    o = (__bf16)val;
  }
  Xin[idx] = o;
}

// w_embed [24][256] -> wembT [256 n][64 k] (zero-padded K)
__global__ void pack_wembT(const float* __restrict__ src, __bf16* __restrict__ dst)
{
  int idx = blockIdx.x * blockDim.x + threadIdx.x;
  if (idx >= 256 * 64) return;
  int n = idx >> 6, k = idx & 63;
  dst[idx] = (k < 24) ? (__bf16)src[k * 256 + n] : (__bf16)0.f;
}

// head_w [256][24] -> wheadT [32 n][256 k] (zero-padded N)
__global__ void pack_wheadT(const float* __restrict__ src, __bf16* __restrict__ dst)
{
  int idx = blockIdx.x * blockDim.x + threadIdx.x;
  if (idx >= 32 * 256) return;
  int n = idx >> 8, k = idx & 255;
  dst[idx] = (n < 24) ? (__bf16)src[k * 24 + n] : (__bf16)0.f;
}

// ---------------------------------------------------------------------------
static inline void launch_gemm(int mode, bool nguard, dim3 grid, hipStream_t s,
                               const __bf16* A, ull sA, int lda,
                               const __bf16* Bt, ull sB, int ldb,
                               const float* bias, void* C, ull sC, int ldc,
                               int N, int K, int Nlog)
{
#define GO(M, G) gemm_bf16_wmma<M, G><<<grid, 256, 0, s>>>(A, sA, lda, Bt, sB, ldb, bias, C, sC, ldc, N, K, Nlog)
  if (nguard) {
    switch (mode) { case 0: GO(0, true); break; case 1: GO(1, true); break;
                    case 2: GO(2, true); break; default: GO(3, true); break; }
  } else {
    switch (mode) { case 0: GO(0, false); break; case 1: GO(1, false); break;
                    case 2: GO(2, false); break; default: GO(3, false); break; }
  }
#undef GO
}

extern "C" void kernel_launch(void* const* d_in, const int* in_sizes, int n_in,
                              void* d_out, int out_size, void* d_ws, size_t ws_size,
                              hipStream_t stream)
{
  (void)in_sizes; (void)n_in; (void)out_size; (void)ws_size;
  const float* y      = (const float*)d_in[0];
  const float* tp     = (const float*)d_in[1];
  const float* w_emb  = (const float*)d_in[2];
  const float* b_emb  = (const float*)d_in[3];
  const float* ln1_g  = (const float*)d_in[4];
  const float* ln1_b  = (const float*)d_in[5];
  const float* tok_w1 = (const float*)d_in[6];
  const float* tok_b1 = (const float*)d_in[7];
  const float* tok_w2 = (const float*)d_in[8];
  const float* tok_b2 = (const float*)d_in[9];
  const float* ln2_g  = (const float*)d_in[10];
  const float* ln2_b  = (const float*)d_in[11];
  const float* ch_w1  = (const float*)d_in[12];
  const float* ch_b1  = (const float*)d_in[13];
  const float* ch_w2  = (const float*)d_in[14];
  const float* ch_b2  = (const float*)d_in[15];
  const float* lnf_g  = (const float*)d_in[16];
  const float* lnf_b  = (const float*)d_in[17];
  const float* head_w = (const float*)d_in[18];
  const float* head_b = (const float*)d_in[19];
  float* out = (float*)d_out;

  const float power_ratio = 1.6f / 0.6f;
  const float pp    = power_ratio / (power_ratio + 1.0f);
  const float scale = pp / (pp * pp + 0.1f);

  // ---- carve workspace ----
  unsigned char* base = (unsigned char*)d_ws;
  size_t off = 0;
  auto carve = [&](size_t bytes) -> void* {
    void* p = base + off;
    off += (bytes + 255) & ~(size_t)255;
    return p;
  };
  float*  x      = (float*) carve((size_t)RTOT * H_DIM * 4);
  __bf16* Xin    = (__bf16*)carve((size_t)RTOT * 64 * 2);
  __bf16* lnbuf  = (__bf16*)carve((size_t)RTOT * H_DIM * 2);
  __bf16* act1   = (__bf16*)carve((size_t)B_DIM * H_DIM * TM_DIM * 2);
  __bf16* act2   = (__bf16*)carve((size_t)CHM * CM_DIM * 2);
  __bf16* wt1T   = (__bf16*)carve((size_t)TM_DIM * NP_DIM * 2);
  __bf16* wt2T   = (__bf16*)carve((size_t)NP_DIM * TM_DIM * 2);
  __bf16* wc1T   = (__bf16*)carve((size_t)CM_DIM * H_DIM * 2);
  __bf16* wc2T   = (__bf16*)carve((size_t)H_DIM * CM_DIM * 2);
  __bf16* wembT  = (__bf16*)carve((size_t)H_DIM * 64 * 2);
  __bf16* wheadT = (__bf16*)carve((size_t)32 * H_DIM * 2);

  // ---- embed ----
  pack_embed_input<<<CDIV(RTOT * 64, 256), 256, 0, stream>>>(y, tp, Xin, scale);
  pack_wembT<<<CDIV(256 * 64, 256), 256, 0, stream>>>(w_emb, wembT);
  pack_wheadT<<<CDIV(32 * 256, 256), 256, 0, stream>>>(head_w, wheadT);

  launch_gemm(EPI_F32_BIAS, false, dim3(CDIV(H_DIM, BN), RTOT / BM, 1), stream,
              Xin, 0ull, 64, wembT, 0ull, 64, b_emb,
              x, 0ull, H_DIM, H_DIM, 64, H_DIM);

  for (int i = 0; i < L_DIM; ++i) {
    convert_transpose_bf16<<<dim3(TM_DIM / 32, NP_DIM / 32), 256, 0, stream>>>(
        tok_w1 + (size_t)i * NP_DIM * TM_DIM, wt1T, NP_DIM, TM_DIM);
    convert_transpose_bf16<<<dim3(NP_DIM / 32, TM_DIM / 32), 256, 0, stream>>>(
        tok_w2 + (size_t)i * TM_DIM * NP_DIM, wt2T, TM_DIM, NP_DIM);
    convert_transpose_bf16<<<dim3(CM_DIM / 32, H_DIM / 32), 256, 0, stream>>>(
        ch_w1 + (size_t)i * H_DIM * CM_DIM, wc1T, H_DIM, CM_DIM);
    convert_transpose_bf16<<<dim3(H_DIM / 32, CM_DIM / 32), 256, 0, stream>>>(
        ch_w2 + (size_t)i * CM_DIM * H_DIM, wc2T, CM_DIM, H_DIM);

    // ---- token mixing ----
    layernorm_bf16<<<RTOT / 8, 256, 0, stream>>>(x, ln1_g + i * H_DIM, ln1_b + i * H_DIM,
                                                 lnbuf, RTOT, 1 /*transposed*/);
    launch_gemm(EPI_BF16_GELU, false, dim3(TM_DIM / BN, H_DIM / BM, B_DIM), stream,
                lnbuf, (ull)H_DIM * NP_DIM, NP_DIM,
                wt1T, 0ull, NP_DIM, tok_b1 + (size_t)i * TM_DIM,
                act1, (ull)H_DIM * TM_DIM, TM_DIM, TM_DIM, NP_DIM, TM_DIM);
    launch_gemm(EPI_TRANS_ACCUM, true, dim3(CDIV(NP_DIM, BN), H_DIM / BM, B_DIM), stream,
                act1, (ull)H_DIM * TM_DIM, TM_DIM,
                wt2T, 0ull, TM_DIM, tok_b2 + (size_t)i * NP_DIM,
                x, (ull)NP_DIM * H_DIM, H_DIM, NP_DIM, TM_DIM, NP_DIM);

    // ---- channel mixing (row-chunked) ----
    layernorm_bf16<<<RTOT / 8, 256, 0, stream>>>(x, ln2_g + i * H_DIM, ln2_b + i * H_DIM,
                                                 lnbuf, RTOT, 0);
    for (int ck = 0; ck < 8; ++ck) {
      size_t ro = (size_t)ck * CHM;
      launch_gemm(EPI_BF16_GELU, false, dim3(CM_DIM / BN, CHM / BM, 1), stream,
                  lnbuf + ro * H_DIM, 0ull, H_DIM,
                  wc1T, 0ull, H_DIM, ch_b1 + (size_t)i * CM_DIM,
                  act2, 0ull, CM_DIM, CM_DIM, H_DIM, CM_DIM);
      launch_gemm(EPI_F32_ACCUM, false, dim3(H_DIM / BN, CHM / BM, 1), stream,
                  act2, 0ull, CM_DIM,
                  wc2T, 0ull, CM_DIM, ch_b2 + (size_t)i * H_DIM,
                  x + ro * H_DIM, 0ull, H_DIM, H_DIM, CM_DIM, H_DIM);
    }
  }

  // ---- final LN + head ----
  layernorm_bf16<<<RTOT / 8, 256, 0, stream>>>(x, lnf_g, lnf_b, lnbuf, RTOT, 0);
  launch_gemm(EPI_F32_BIAS, true, dim3(1, RTOT / BM, 1), stream,
              lnbuf, 0ull, H_DIM, wheadT, 0ull, H_DIM, head_b,
              out, 0ull, 24, 32, H_DIM, 24);
}
